// OurGCN_90666759618859
// MI455X (gfx1250) — compile-verified, matrix operand order checked by reading, not run on hardware
//
#include <hip/hip_runtime.h>
#include <hip/hip_bf16.h>
#include <math.h>

typedef __attribute__((ext_vector_type(16))) __bf16 v16bf;
typedef __attribute__((ext_vector_type(8)))  __bf16 v8bf;
typedef __attribute__((ext_vector_type(8)))  float  v8f;

#define F_IN  512
#define F_HID 256
#define F_OUT 40
#define F_OUTP 48   // padded to 3 WMMA tiles

union BF16x16 { v16bf v; v8bf h[2]; };

// ---------------- degree / norm ----------------
__global__ void gcn_deg_init(float* deg, int n) {
  int i = blockIdx.x * blockDim.x + threadIdx.x;
  if (i < n) deg[i] = 1.0f;  // self-loop contributes 1
}

__global__ void gcn_deg_count(const int* __restrict__ col, float* deg, int e) {
  int i = blockIdx.x * blockDim.x + threadIdx.x;
  if (i < e) atomicAdd(&deg[col[i]], 1.0f);
}

__global__ void gcn_deg_rsqrt(float* deg, int n) {
  int i = blockIdx.x * blockDim.x + threadIdx.x;
  if (i < n) deg[i] = rsqrtf(deg[i]);  // deg >= 1 always
}

// ---------------- weight prep: transpose + bf16 ----------------
// wt1[n*512 + k] = bf16(W1[k,n]),  n<256, k<512
__global__ void gcn_prep_w1t(const float* __restrict__ W1, __bf16* __restrict__ wt1) {
  int idx = blockIdx.x * blockDim.x + threadIdx.x;
  if (idx >= F_HID * F_IN) return;
  int n = idx >> 9, k = idx & (F_IN - 1);
  wt1[idx] = (__bf16)W1[k * F_HID + n];
}

// wt2[n*256 + k] = bf16(W2[k,n]) for n<40, 0 for 40<=n<48
__global__ void gcn_prep_w2t(const float* __restrict__ W2, __bf16* __restrict__ wt2) {
  int idx = blockIdx.x * blockDim.x + threadIdx.x;
  if (idx >= F_OUTP * F_HID) return;
  int n = idx >> 8, k = idx & (F_HID - 1);
  wt2[idx] = (n < F_OUT) ? (__bf16)W2[k * F_OUT + n] : (__bf16)0.0f;
}

// ---------------- GEMM1: h = x @ W1, p1 = dinv^2*h + b1 fused -------------
// One wave owns a full 16-row x 256-col strip: x read exactly once,
// A fragment reused by 16 WMMAs per K-step. grid ceil(mtiles/8), block 256.
__global__ __launch_bounds__(256) void gcn_gemm1(
    const float* __restrict__ x, const __bf16* __restrict__ wt1,
    const float* __restrict__ dinv, const float* __restrict__ b1,
    float* __restrict__ h, float* __restrict__ p1, int mtiles) {
  int wave = threadIdx.x >> 5;
  int lane = threadIdx.x & 31;
  int rowTile = blockIdx.x * 8 + wave;
  if (rowTile >= mtiles) return;               // wave-uniform exit
  int l16 = lane & 15;
  int kb  = (lane >> 4) << 3;                  // 0 or 8
  int m   = (rowTile << 4) + l16;

  v8f acc[16] = {};
  for (int k0 = 0; k0 < F_IN; k0 += 32) {
    // A fragment: row m, K = k0+kb+{0..7} and k0+16+kb+{0..7}
    v16bf a;
    const float* xr = x + (size_t)m * F_IN + k0 + kb;
    #pragma unroll
    for (int j = 0; j < 8; ++j) a[j] = (__bf16)xr[j];
    #pragma unroll
    for (int j = 0; j < 8; ++j) a[j + 8] = (__bf16)xr[16 + j];

    #pragma unroll
    for (int t = 0; t < 16; ++t) {
      int n = (t << 4) + l16;
      const v8bf* wp = (const v8bf*)(wt1 + (size_t)n * F_IN + k0 + kb);
      BF16x16 b;
      b.h[0] = wp[0];   // K = k0+kb ..  +7
      b.h[1] = wp[2];   // K = k0+kb+16..+23
      acc[t] = __builtin_amdgcn_wmma_f32_16x16x32_bf16(
          false, a, false, b.v, (short)0, acc[t], false, false);
    }
  }

  // fused epilogue: h = acc; p1 = dinv^2 * acc + b1
  float dd[8];
  #pragma unroll
  for (int v = 0; v < 8; ++v) {
    float d = dinv[(rowTile << 4) + v + ((lane >> 4) << 3)];
    dd[v] = d * d;
  }
  #pragma unroll
  for (int t = 0; t < 16; ++t) {
    int n = (t << 4) + l16;
    float bn = b1[n];
    #pragma unroll
    for (int v = 0; v < 8; ++v) {
      int mm = (rowTile << 4) + v + ((lane >> 4) << 3);
      float val = acc[t][v];
      h [(size_t)mm * F_HID + n] = val;
      p1[(size_t)mm * F_HID + n] = dd[v] * val + bn;
    }
  }
}

// ---------------- scatter, 256 feats: one wave per edge ----------------
__global__ __launch_bounds__(256) void gcn_scatter256(
    const int* __restrict__ row, const int* __restrict__ col,
    const float* __restrict__ dinv, const float* __restrict__ h,
    float* __restrict__ p, int e) {
  int wid = (int)(((long long)blockIdx.x * blockDim.x + threadIdx.x) >> 5);
  if (wid >= e) return;
  int lane = threadIdx.x & 31;
  int r = row[wid], c = col[wid];
  float nrm = dinv[r] * dinv[c];
  const float4* hs = (const float4*)(h + (size_t)c * F_HID);
  float4 A = hs[lane];        // feats lane*4 .. lane*4+3
  float4 B = hs[lane + 32];   // feats 128+lane*4 ..
  float* dst = p + (size_t)r * F_HID;
  int f0 = lane << 2;
  atomicAdd(dst + f0 + 0, nrm * A.x);
  atomicAdd(dst + f0 + 1, nrm * A.y);
  atomicAdd(dst + f0 + 2, nrm * A.z);
  atomicAdd(dst + f0 + 3, nrm * A.w);
  atomicAdd(dst + 128 + f0 + 0, nrm * B.x);
  atomicAdd(dst + 128 + f0 + 1, nrm * B.y);
  atomicAdd(dst + 128 + f0 + 2, nrm * B.z);
  atomicAdd(dst + 128 + f0 + 3, nrm * B.w);
}

// ------ GEMM2: h2 = relu(p1) @ W2, out = dinv^2*h2 + b2 fused -----------
__global__ __launch_bounds__(256) void gcn_gemm2(
    const float* __restrict__ p1, const __bf16* __restrict__ wt2,
    const float* __restrict__ dinv, const float* __restrict__ b2,
    float* __restrict__ h2, float* __restrict__ out, int mtiles) {
  int wave = threadIdx.x >> 5;
  int lane = threadIdx.x & 31;
  int rowTile = blockIdx.x * 8 + wave;
  if (rowTile >= mtiles) return;               // wave-uniform exit
  int l16 = lane & 15;
  int kb  = (lane >> 4) << 3;
  int m   = (rowTile << 4) + l16;

  v8f acc[3] = {};
  for (int k0 = 0; k0 < F_HID; k0 += 32) {
    v16bf a;  // ReLU fused into A load
    const float* pr = p1 + (size_t)m * F_HID + k0 + kb;
    #pragma unroll
    for (int j = 0; j < 8; ++j) a[j]     = (__bf16)fmaxf(pr[j], 0.0f);
    #pragma unroll
    for (int j = 0; j < 8; ++j) a[j + 8] = (__bf16)fmaxf(pr[16 + j], 0.0f);

    #pragma unroll
    for (int t = 0; t < 3; ++t) {
      int n = (t << 4) + l16;
      const v8bf* wp = (const v8bf*)(wt2 + (size_t)n * F_HID + k0 + kb);
      BF16x16 b;
      b.h[0] = wp[0];
      b.h[1] = wp[2];
      acc[t] = __builtin_amdgcn_wmma_f32_16x16x32_bf16(
          false, a, false, b.v, (short)0, acc[t], false, false);
    }
  }

  float dd[8];
  #pragma unroll
  for (int v = 0; v < 8; ++v) {
    float d = dinv[(rowTile << 4) + v + ((lane >> 4) << 3)];
    dd[v] = d * d;
  }
  #pragma unroll
  for (int t = 0; t < 3; ++t) {
    int n = (t << 4) + l16;
    if (n < F_OUT) {
      float bn = b2[n];
      #pragma unroll
      for (int v = 0; v < 8; ++v) {
        int mm = (rowTile << 4) + v + ((lane >> 4) << 3);
        float val = acc[t][v];
        h2 [(size_t)mm * F_OUT + n] = val;
        out[(size_t)mm * F_OUT + n] = dd[v] * val + bn;
      }
    }
  }
}

// ---------------- scatter, 40 feats: one wave per edge ----------------
__global__ __launch_bounds__(256) void gcn_scatter40(
    const int* __restrict__ row, const int* __restrict__ col,
    const float* __restrict__ dinv, const float* __restrict__ h2,
    float* __restrict__ out, int e) {
  int wid = (int)(((long long)blockIdx.x * blockDim.x + threadIdx.x) >> 5);
  if (wid >= e) return;
  int lane = threadIdx.x & 31;
  int r = row[wid], c = col[wid];
  float nrm = dinv[r] * dinv[c];
  const float* src = h2 + (size_t)c * F_OUT;
  float* dst = out + (size_t)r * F_OUT;
  for (int f = lane; f < F_OUT; f += 32)
    atomicAdd(dst + f, nrm * src[f]);
}

// ---------------- log_softmax: one wave per row ----------------
__global__ __launch_bounds__(256) void gcn_logsoftmax(
    const float* __restrict__ fin, float* __restrict__ ls, int n) {
  int r = blockIdx.x * 8 + (threadIdx.x >> 5);
  if (r >= n) return;
  int lane = threadIdx.x & 31;
  const float* row = fin + (size_t)r * F_OUT;
  float v0 = row[lane];                                   // lane < 32 < 40
  float v1 = (lane + 32 < F_OUT) ? row[lane + 32] : -INFINITY;
  float mx = fmaxf(v0, v1);
  #pragma unroll
  for (int off = 16; off > 0; off >>= 1)
    mx = fmaxf(mx, __shfl_xor(mx, off, 32));
  float s = expf(v0 - mx) + ((lane + 32 < F_OUT) ? expf(v1 - mx) : 0.0f);
  #pragma unroll
  for (int off = 16; off > 0; off >>= 1)
    s += __shfl_xor(s, off, 32);
  float lse = mx + logf(s);
  float* o = ls + (size_t)r * F_OUT;
  o[lane] = v0 - lse;
  if (lane + 32 < F_OUT) o[lane + 32] = v1 - lse;
}

extern "C" void kernel_launch(void* const* d_in, const int* in_sizes, int n_in,
                              void* d_out, int out_size, void* d_ws, size_t ws_size,
                              hipStream_t stream) {
  const float* x  = (const float*)d_in[0];
  const int*   ei = (const int*)d_in[1];
  const float* W1 = (const float*)d_in[2];
  const float* b1 = (const float*)d_in[3];
  const float* W2 = (const float*)d_in[4];
  const float* b2 = (const float*)d_in[5];
  float* out = (float*)d_out;

  const int N = in_sizes[0] / F_IN;       // 100000
  const int E = in_sizes[1] / 2;          // 3200000
  const int* row = ei;
  const int* col = ei + E;

  // workspace layout (all 16B-aligned for the sizes at hand)
  char* ws = (char*)d_ws;
  float* dinv = (float*)ws;                               // N
  float* h    = (float*)(ws + (size_t)N * 4);             // N x 256
  float* p1   = h  + (size_t)N * F_HID;                   // N x 256
  float* h2   = p1 + (size_t)N * F_HID;                   // N x 40
  __bf16* wt1 = (__bf16*)(h2 + (size_t)N * F_OUT);        // 256 x 512
  __bf16* wt2 = wt1 + (size_t)F_HID * F_IN;               // 48 x 256

  const int mtiles = (N + 15) / 16;                       // 6250 (exact)
  const int gBlocks = (mtiles + 7) / 8;                   // 782
  const int eWaveBlocks = (E + 7) / 8;                    // 8 edge-waves / block

  // 1) norm
  gcn_deg_init <<<(N + 255) / 256, 256, 0, stream>>>(dinv, N);
  gcn_deg_count<<<(E + 255) / 256, 256, 0, stream>>>(col, dinv, E);
  gcn_deg_rsqrt<<<(N + 255) / 256, 256, 0, stream>>>(dinv, N);

  // 2) weight prep (transpose + bf16 so B fragments are contiguous b128 loads)
  gcn_prep_w1t<<<(F_HID * F_IN + 255) / 256, 256, 0, stream>>>(W1, wt1);
  gcn_prep_w2t<<<(F_OUTP * F_HID + 255) / 256, 256, 0, stream>>>(W2, wt2);

  // 3) layer 1: GEMM (+fused self-loop/bias init of p1), then edge scatter
  gcn_gemm1<<<gBlocks, 256, 0, stream>>>(x, wt1, dinv, b1, h, p1, mtiles);
  gcn_scatter256<<<eWaveBlocks, 256, 0, stream>>>(row, col, dinv, h, p1, E);

  // 4) layer 2: GEMM (ReLU fused on A, fused init of out), then edge scatter
  gcn_gemm2<<<gBlocks, 256, 0, stream>>>(p1, wt2, dinv, b2, h2, out, mtiles);
  gcn_scatter40<<<eWaveBlocks, 256, 0, stream>>>(row, col, dinv, h2, out, E);

  // 5) log_softmax into second half of d_out
  gcn_logsoftmax<<<(N + 7) / 8, 256, 0, stream>>>(out, out + (size_t)N * F_OUT, N);
}
